// Gaussian_MultiHeadSA_63213328662850
// MI455X (gfx1250) — matrix operand order repaired and checked by reading, not compiled, not run on hardware
//
#include <hip/hip_runtime.h>

// ---------------- problem constants ----------------
#define Bd   8
#define Td   2048
#define Dd   256
#define Ed   4
#define DEd  64
#define BTd  (Bd * Td)

typedef _Float16 h16;
typedef __attribute__((ext_vector_type(16))) _Float16 v16h;
typedef __attribute__((ext_vector_type(8)))  _Float16 v8h;
typedef __attribute__((ext_vector_type(8)))  float    v8f;

union AFrag { v16h v; v8h h[2]; };

__device__ __forceinline__ v8f wmma_f32_16x16x32_f16(v16h a, v16h b, v8f c) {
  // 8 args: (neg_a, A, neg_b, B, c_mod, C, reuse_a, reuse_b)
  return __builtin_amdgcn_wmma_f32_16x16x32_f16(false, a, false, b, (short)0, c,
                                                false, false);
}

// CDNA5 async DMA: global -> LDS, 16B per lane, tracked by ASYNCcnt.
// LDS address comes from the vdst VGPR; global = SGPR_base + VGPR_offset (GVS mode).
__device__ __forceinline__ void async_copy_b128(unsigned lds_off,
                                                const void* base,
                                                unsigned byte_off) {
  asm volatile("global_load_async_to_lds_b128 %0, %1, %2"
               :
               : "v"(lds_off), "v"(byte_off), "s"(base)
               : "memory");
}

__device__ __forceinline__ void wait_asynccnt0() {
  asm volatile("s_wait_asynccnt 0x0" ::: "memory");
}

// ---------------- fp32 -> fp16 conversion ----------------
__global__ __launch_bounds__(256) void cvt_f32_f16_kernel(
    const float* __restrict__ src, h16* __restrict__ dst, int n) {
  int i = blockIdx.x * blockDim.x + threadIdx.x;
  if (i < n) dst[i] = (h16)src[i];
}

// ---------------- QKV projection: Y = x @ W^T + b, scatter to head layout --
// Grid: (D/64, BT/16), block: 32 (one wave = one 16x64 output tile, 4 accs
// sharing each A fragment -> 4 WMMA per k-step, 32 per wave).
// Epilogue inverts the reference's scrambled _split_heads:
//   token j, feature i -> bh = b*4 + (i>>6), t = ((i&63)<<5)|(j>>6), d = j&63
// With 64-wide feature tiles, i>>6 == nT (constant head per block).
// Q/K: masked by (t < seq_len[b]), stored [bh][t][d].  V: stored transposed
// [bh][d][t] so the attention PV B-operand is a contiguous row load.
__global__ __launch_bounds__(32) void qkv_gemm_kernel(
    const h16* __restrict__ xh, const h16* __restrict__ wh,
    const float* __restrict__ bias, const int* __restrict__ seq_len,
    h16* __restrict__ dst, int is_v) {
  const int lane = threadIdx.x & 31;
  const int hi   = lane >> 4;
  const int lo   = lane & 15;
  const int nT   = blockIdx.x;       // 64-wide feature tile (0..3) == head
  const int mT   = blockIdx.y;       // global token tile (0..1023)

  const h16* arow = xh + (size_t)(mT * 16 + lo) * Dd;    // A: row m = lo
  const h16* brow = wh + (size_t)(nT * 64 + lo) * Dd;    // B rows nT*64+c*16+lo

  v8f zero = {};
  v8f acc[4] = {zero, zero, zero, zero};
  #pragma unroll
  for (int ks = 0; ks < Dd; ks += 32) {
    AFrag a;
    a.v = *(const v16h*)(arow + ks + 16 * hi);
    #pragma unroll
    for (int c = 0; c < 4; ++c) {
      v16h b = *(const v16h*)(brow + (size_t)c * 16 * Dd + ks + 16 * hi);
      acc[c] = wmma_f32_16x16x32_f16(a.v, b, acc[c]);
    }
  }

  const int b  = mT >> 7;                 // T/16 = 128 tiles per batch
  const int bh = (b << 2) + nT;
  const int sl = seq_len[b];

  #pragma unroll
  for (int c = 0; c < 4; ++c) {
    const int i = nT * 64 + c * 16 + lo;  // feature column for this lane
    const float bv = bias[i];
    #pragma unroll
    for (int r = 0; r < 8; ++r) {
      const int j = (mT & 127) * 16 + r + 8 * hi;   // token within batch
      const float y = acc[c][r] + bv;
      const int t = ((i & 63) << 5) | (j >> 6);
      const int d = j & 63;
      if (is_v) {
        dst[((size_t)bh * DEd + d) * Td + t] = (h16)y;        // Vt[bh][d][t]
      } else {
        const float msk = (t < sl) ? 1.0f : 0.0f;
        dst[((size_t)bh * Td + t) * DEd + d] = (h16)(y * msk); // Qh/Kh
      }
    }
  }
}

// ---------------- flash attention with Gaussian positional punish ----------
// Grid: (T/64, B*E), block: 128 (4 waves; each wave owns one 16-row Q tile;
// all 4 waves share the same (b,h)).  K/V tiles for each 32-key step are
// staged into LDS once per block with global_load_async_to_lds_b128 (double
// buffered, ASYNCcnt + one barrier per step), then consumed 4x from LDS.
// Per step/wave: 4 WMMA for S = Q K^T, punish * scale, ==0 -> -2e20, online
// softmax, P staged through LDS (C-layout -> A-layout), 4 WMMA for ctx += PV.
__global__ __launch_bounds__(128) void attn_kernel(
    const h16* __restrict__ Qh, const h16* __restrict__ Kh,
    const h16* __restrict__ Vt, h16* __restrict__ ctxM,
    const float* __restrict__ theta) {
  const int tid  = threadIdx.x;
  const int wave = tid >> 5;
  const int lane = tid & 31;
  const int hi   = lane >> 4;
  const int lo   = lane & 15;
  const int bh   = blockIdx.y;
  const int b    = bh >> 2;
  const int h    = bh & 3;
  const int q0   = (blockIdx.x * 4 + wave) * 16;

  __shared__ __align__(32) _Float16 kbuf[2][32 * DEd];   // 2 x 4KB, rows j
  __shared__ __align__(32) _Float16 vbuf[2][DEd * 32];   // 2 x 4KB, rows d
  __shared__ __align__(32) _Float16 plds[4][16 * 32];    // per-wave P tile
  _Float16* pbuf = plds[wave];

  // Q A-operands (K-dim 64 -> two 16x32 fragments), resident for whole loop.
  const h16* qrow = Qh + ((size_t)bh * Td + q0 + lo) * DEd;
  AFrag qa0, qa1;
  qa0.v = *(const v16h*)(qrow + 16 * hi);
  qa1.v = *(const v16h*)(qrow + 32 + 16 * hi);

  const float th      = theta[0];
  const float inv_t2  = 1.0f / (th * th);
  const float inv_sqT = 0.022097086912079608f;   // 1/sqrt(2048)

  float mst[8], lst[8];
  #pragma unroll
  for (int r = 0; r < 8; ++r) { mst[r] = -3.0e38f; lst[r] = 0.0f; }
  v8f zero = {};
  v8f acc[4] = {zero, zero, zero, zero};

  // ---- async stage of one 32-key K tile + V tile into LDS buffer nb ----
  // K tile: 4KB contiguous at Kh + (bh*T + j0)*64 halves.
  // V tile: 64 rows of 64B from Vt[bh][d][j0..j0+31].
  auto stage = [&](int nb, int j0) {
    const unsigned kl = (unsigned)(uintptr_t)&kbuf[nb][0];
    const unsigned vl = (unsigned)(uintptr_t)&vbuf[nb][0];
    const unsigned kg = (unsigned)(((size_t)bh * Td + j0) * DEd * 2);
    #pragma unroll
    for (int c = tid; c < 256; c += 128) {        // 2 x 16B chunks each
      async_copy_b128(kl + c * 16, Kh, kg + c * 16);
      const int d  = c >> 2;
      const int cc = c & 3;
      const unsigned vg =
          (unsigned)((((size_t)bh * DEd + d) * Td + j0) * 2) + cc * 16;
      async_copy_b128(vl + d * 64 + cc * 16, Vt, vg);
    }
  };

  stage(0, 0);

  for (int j0 = 0; j0 < Td; j0 += 32) {
    const int cur = (j0 >> 5) & 1;
    wait_asynccnt0();      // my async writes into buf `cur` are done
    __syncthreads();       // everyone's writes done; everyone done reading ^cur
    if (j0 + 32 < Td) stage(cur ^ 1, j0 + 32);

    const h16* kb = &kbuf[cur][0];
    const h16* vb = &vbuf[cur][0];

    // ---- S = Q K^T for 32 key rows (two 16-col sub-tiles), from LDS ----
    v8f s0 = {}, s1 = {};
    {
      v16h b0 = *(const v16h*)(kb + lo * DEd + 16 * hi);
      v16h b1 = *(const v16h*)(kb + lo * DEd + 32 + 16 * hi);
      s0 = wmma_f32_16x16x32_f16(qa0.v, b0, s0);
      s0 = wmma_f32_16x16x32_f16(qa1.v, b1, s0);
      v16h b2 = *(const v16h*)(kb + (16 + lo) * DEd + 16 * hi);
      v16h b3 = *(const v16h*)(kb + (16 + lo) * DEd + 32 + 16 * hi);
      s1 = wmma_f32_16x16x32_f16(qa0.v, b2, s1);
      s1 = wmma_f32_16x16x32_f16(qa1.v, b3, s1);
    }

    // ---- scale, Gaussian punish, exact-zero -> -2e20, row max ----
    float rowmax[8];
    #pragma unroll
    for (int r = 0; r < 8; ++r) {
      const int m = r + 8 * hi;
      const float d0 = (float)((j0 + lo) - (q0 + m));      // key - query
      const float d1 = d0 + 16.0f;
      const float pz0 = __expf(-d0 * d0 * inv_t2);
      const float pz1 = __expf(-d1 * d1 * inv_t2);
      float v0 = s0[r] * inv_sqT * pz0;
      float v1 = s1[r] * inv_sqT * pz1;
      v0 = (v0 == 0.0f) ? -2.0e20f : v0;
      v1 = (v1 == 0.0f) ? -2.0e20f : v1;
      s0[r] = v0; s1[r] = v1;
      rowmax[r] = fmaxf(v0, v1);
    }
    #pragma unroll
    for (int r = 0; r < 8; ++r)
      #pragma unroll
      for (int off = 8; off; off >>= 1)
        rowmax[r] = fmaxf(rowmax[r], __shfl_xor(rowmax[r], off, 32));

    // ---- online softmax update ----
    float corr[8];
    #pragma unroll
    for (int r = 0; r < 8; ++r) {
      const float mn = fmaxf(mst[r], rowmax[r]);
      corr[r] = __expf(mst[r] - mn);
      mst[r] = mn;
      s0[r] = __expf(s0[r] - mn);
      s1[r] = __expf(s1[r] - mn);
    }
    #pragma unroll
    for (int r = 0; r < 8; ++r) {
      float ps = s0[r] + s1[r];
      #pragma unroll
      for (int off = 8; off; off >>= 1) ps += __shfl_xor(ps, off, 32);
      lst[r] = lst[r] * corr[r] + ps;
    }
    #pragma unroll
    for (int c = 0; c < 4; ++c)
      #pragma unroll
      for (int r = 0; r < 8; ++r) acc[c][r] *= corr[r];

    // ---- P: C-layout -> A-layout via per-wave LDS tile (16x32 fp16) ----
    #pragma unroll
    for (int r = 0; r < 8; ++r) {
      const int m = r + 8 * hi;
      pbuf[m * 32 + lo]      = (h16)s0[r];
      pbuf[m * 32 + 16 + lo] = (h16)s1[r];
    }
    AFrag pa;
    pa.v = *(const v16h*)(pbuf + lo * 32 + 16 * hi);

    // ---- ctx += P V over the 64-wide head dim (4 n-chunks), V from LDS ----
    #pragma unroll
    for (int c = 0; c < 4; ++c) {
      v16h vv = *(const v16h*)(vb + (c * 16 + lo) * 32 + 16 * hi);
      acc[c] = wmma_f32_16x16x32_f16(pa.v, vv, acc[c]);
    }
  }

  // ---- normalize and write merged ctx: ctxM[b][t][h*64 + d] (fp16) ----
  float invl[8];
  #pragma unroll
  for (int r = 0; r < 8; ++r) invl[r] = 1.0f / lst[r];
  const size_t rowbase = (size_t)b * Td + q0;
  #pragma unroll
  for (int c = 0; c < 4; ++c) {
    const int col = h * 64 + c * 16 + lo;
    #pragma unroll
    for (int r = 0; r < 8; ++r) {
      const int m = r + 8 * hi;
      ctxM[(rowbase + m) * Dd + col] = (h16)(acc[c][r] * invl[r]);
    }
  }
}

// ---------------- output projection: out = ctxM @ wo^T + bo + x -----------
// Grid: (D/64, BT/16), block: 32; 16x64 tile per wave, 32 WMMA.
__global__ __launch_bounds__(32) void out_gemm_kernel(
    const h16* __restrict__ ctxM, const h16* __restrict__ woh,
    const float* __restrict__ bo, const float* __restrict__ x,
    float* __restrict__ out) {
  const int lane = threadIdx.x & 31;
  const int hi   = lane >> 4;
  const int lo   = lane & 15;
  const int nT   = blockIdx.x;
  const int mT   = blockIdx.y;

  const h16* arow = ctxM + (size_t)(mT * 16 + lo) * Dd;
  const h16* brow = woh  + (size_t)(nT * 64 + lo) * Dd;

  v8f zero = {};
  v8f acc[4] = {zero, zero, zero, zero};
  #pragma unroll
  for (int ks = 0; ks < Dd; ks += 32) {
    AFrag a;
    a.v = *(const v16h*)(arow + ks + 16 * hi);
    #pragma unroll
    for (int c = 0; c < 4; ++c) {
      v16h b = *(const v16h*)(brow + (size_t)c * 16 * Dd + ks + 16 * hi);
      acc[c] = wmma_f32_16x16x32_f16(a.v, b, acc[c]);
    }
  }

  #pragma unroll
  for (int c = 0; c < 4; ++c) {
    const int i = nT * 64 + c * 16 + lo;
    const float bv = bo[i];
    #pragma unroll
    for (int r = 0; r < 8; ++r) {
      const size_t row = (size_t)mT * 16 + r + 8 * hi;   // global (b*T + t)
      out[row * Dd + i] = acc[c][r] + bv + x[row * Dd + i];
    }
  }
}

// ---------------- launch ----------------
extern "C" void kernel_launch(void* const* d_in, const int* in_sizes, int n_in,
                              void* d_out, int out_size, void* d_ws, size_t ws_size,
                              hipStream_t stream) {
  const float* x       = (const float*)d_in[0];
  const int*   seq_len = (const int*)d_in[1];
  const float* wq = (const float*)d_in[2];  const float* bq = (const float*)d_in[3];
  const float* wk = (const float*)d_in[4];  const float* bk = (const float*)d_in[5];
  const float* wv = (const float*)d_in[6];  const float* bv = (const float*)d_in[7];
  const float* wo = (const float*)d_in[8];  const float* bo = (const float*)d_in[9];
  const float* theta = (const float*)d_in[10];

  // workspace layout (~42.5 MB, fp16)
  char* ws = (char*)d_ws;
  size_t off = 0;
  h16* xh   = (h16*)(ws + off); off += (size_t)BTd * Dd * 2;
  h16* wqh  = (h16*)(ws + off); off += (size_t)Dd * Dd * 2;
  h16* wkh  = (h16*)(ws + off); off += (size_t)Dd * Dd * 2;
  h16* wvh  = (h16*)(ws + off); off += (size_t)Dd * Dd * 2;
  h16* woh  = (h16*)(ws + off); off += (size_t)Dd * Dd * 2;
  h16* Qh   = (h16*)(ws + off); off += (size_t)BTd * Dd * 2;   // [32][2048][64]
  h16* Kh   = (h16*)(ws + off); off += (size_t)BTd * Dd * 2;   // [32][2048][64]
  h16* Vt   = (h16*)(ws + off); off += (size_t)BTd * Dd * 2;   // [32][64][2048]
  h16* ctxM = (h16*)(ws + off); off += (size_t)BTd * Dd * 2;   // [B][T][D]

  // fp32 -> fp16 staging
  cvt_f32_f16_kernel<<<(BTd * Dd + 255) / 256, 256, 0, stream>>>(x, xh, BTd * Dd);
  cvt_f32_f16_kernel<<<(Dd * Dd + 255) / 256, 256, 0, stream>>>(wq, wqh, Dd * Dd);
  cvt_f32_f16_kernel<<<(Dd * Dd + 255) / 256, 256, 0, stream>>>(wk, wkh, Dd * Dd);
  cvt_f32_f16_kernel<<<(Dd * Dd + 255) / 256, 256, 0, stream>>>(wv, wvh, Dd * Dd);
  cvt_f32_f16_kernel<<<(Dd * Dd + 255) / 256, 256, 0, stream>>>(wo, woh, Dd * Dd);

  // QKV projections with scrambled split-heads + mask (Q,K) / transpose (V)
  dim3 gGemm(Dd / 64, BTd / 16);
  qkv_gemm_kernel<<<gGemm, 32, 0, stream>>>(xh, wqh, bq, seq_len, Qh, 0);
  qkv_gemm_kernel<<<gGemm, 32, 0, stream>>>(xh, wkh, bk, seq_len, Kh, 0);
  qkv_gemm_kernel<<<gGemm, 32, 0, stream>>>(xh, wvh, bv, seq_len, Vt, 1);

  // flash attention with Gaussian punish (async-LDS staged K/V)
  attn_kernel<<<dim3(Td / 64, Bd * Ed), 128, 0, stream>>>(Qh, Kh, Vt, ctxM, theta);

  // output projection + bias + residual
  out_gemm_kernel<<<gGemm, 32, 0, stream>>>(ctxM, woh, bo, x, (float*)d_out);
}
